// VLMWithSpatialActionExpertModel_84198538871018
// MI455X (gfx1250) — matrix operand order repaired and checked by reading, not compiled, not run on hardware
//
#include <hip/hip_runtime.h>
#include <cstdint>
#include <cstddef>

// ---------- types ----------
typedef __attribute__((ext_vector_type(16))) __bf16 bf16x16;
typedef __attribute__((ext_vector_type(8)))  float  f32x8;
typedef __attribute__((ext_vector_type(4)))  unsigned int u32x4v;
typedef __attribute__((ext_vector_type(8)))  int i32x8v;
typedef __attribute__((ext_vector_type(4)))  int i32x4v;

struct alignas(16) UInt4x { unsigned int x, y, z, w; };
struct alignas(16) Frag2x { UInt4x lo, hi; };

#if defined(__has_builtin)
#  if __has_builtin(__builtin_amdgcn_tensor_load_to_lds)
#    define USE_TDM 1
#  endif
#endif
#ifndef USE_TDM
#  define USE_TDM 0
#endif

__device__ __forceinline__ f32x8 zero8() {
  f32x8 z = {0.f,0.f,0.f,0.f,0.f,0.f,0.f,0.f};
  return z;
}
__device__ __forceinline__ unsigned int f2bf_u(float f) {
  unsigned int u = __float_as_uint(f);
  u += 0x7fffu + ((u >> 16) & 1u);   // round-to-nearest-even
  return u >> 16;
}
__device__ __forceinline__ unsigned short f2bf(float f) { return (unsigned short)f2bf_u(f); }
__device__ __forceinline__ float bf2f(unsigned short u) {
  return __uint_as_float(((unsigned int)u) << 16);
}
__device__ __forceinline__ bf16x16 ld_frag(const unsigned short* lo, const unsigned short* hi) {
  Frag2x f;
  f.lo = *(const UInt4x*)lo;
  f.hi = *(const UInt4x*)hi;
  return __builtin_bit_cast(bf16x16, f);
}
__device__ __forceinline__ f32x8 wmma_bf16(bf16x16 a, bf16x16 b, f32x8 c) {
  return __builtin_amdgcn_wmma_f32_16x16x32_bf16(false, a, false, b, (short)0, c, false, false);
}
__device__ __forceinline__ float gelu_tanh(float x) {
  float x3 = x * x * x;
  return 0.5f * x * (1.0f + tanhf(0.79788456080286535588f * (x + 0.044715f * x3)));
}

// LDS byte offset of a __shared__ object (addrspace(3) pointers are LDS offsets)
__device__ __forceinline__ unsigned lds_off(const void* p) {
  return (unsigned)(unsigned long long)(__attribute__((address_space(3))) const void*)p;
}

#if USE_TDM
// TDM: DMA a 128-row x 32-col bf16 tile (row stride = stride_elems) from global
// into LDS with 32B padding after each 64B row => 48-element padded LDS rows.
__device__ __forceinline__ void tdm_load_tile_128x32(const unsigned short* gaddr,
                                                     unsigned ldsbyte, int stride_elems) {
  unsigned long long ga = (unsigned long long)gaddr;
  u32x4v g0;
  g0[0] = 1u;                                         // count=1, user desc, no gather
  g0[1] = ldsbyte;                                    // lds_addr
  g0[2] = (unsigned)(ga & 0xffffffffu);               // global_addr[31:0]
  g0[3] = (unsigned)((ga >> 32) & 0x1ffffffu)         // global_addr[56:32]
        | 0x80000000u;                                // type=2 (image)
  i32x8v g1;
  g1[0] = (1 << 16)                                   // data_size = 2 bytes
        | (1 << 20)                                   // pad_enable
        | (3 << 22)                                   // pad_interval: 16 DWORDs (64B row)
        | (7 << 25);                                  // pad_amount: 8 DWORDs (32B)
  g1[1] = 0;                                          // dim0 low16 << 16 (dim0 = 1<<30)
  g1[2] = 0x4000;                                     // dim0 high16 | dim1 low16<<16
  g1[3] = 0x4000 | (32 << 16);                        // dim1 high16 | tile_dim0=32
  g1[4] = 128;                                        // tile_dim1=128, tile_dim2=0
  g1[5] = stride_elems;                               // tensor_dim0_stride low32
  g1[6] = 0;
  g1[7] = 0;
  i32x4v z4 = {0, 0, 0, 0};
  i32x8v z8 = {0, 0, 0, 0, 0, 0, 0, 0};
  __builtin_amdgcn_tensor_load_to_lds(g0, g1, z4, z4, z8, 0);
}
#endif

// ---------- f32 -> bf16 bulk convert (weights, once per launch) ----------
__global__ void __launch_bounds__(256)
cvt_kernel(const float* __restrict__ src, unsigned short* __restrict__ dst, size_t n) {
  size_t i = (size_t)blockIdx.x * 256 + threadIdx.x;
  size_t step = (size_t)gridDim.x * 256;
  for (; i < n; i += step) dst[i] = f2bf(src[i]);
}

// ---------- RMSNorm: bf16 out ----------
__global__ void __launch_bounds__(256)
rmsnorm_kernel(const float* __restrict__ x, const float* __restrict__ w,
               unsigned short* __restrict__ out, int M, int D) {
  int row = blockIdx.x;
  if (row >= M) return;
  const float* xr = x + (size_t)row * D;
  float ss = 0.f;
  for (int i = threadIdx.x; i < D; i += 256) { float v = xr[i]; ss += v * v; }
#pragma unroll
  for (int off = 16; off > 0; off >>= 1) ss += __shfl_xor(ss, off, 32);
  __shared__ float red[8];
  if ((threadIdx.x & 31) == 0) red[threadIdx.x >> 5] = ss;
  __syncthreads();
  float tot = red[0]+red[1]+red[2]+red[3]+red[4]+red[5]+red[6]+red[7];
  float inv = rsqrtf(tot / (float)D + 1e-6f);
  unsigned short* orow = out + (size_t)row * D;
  for (int i = threadIdx.x; i < D; i += 256) orow[i] = f2bf(xr[i] * inv * (1.0f + w[i]));
}

// ---------- GEMM: C[M,N] = A[M,K]bf16 @ B[N,K]^T bf16, f32 accum ----------
// EPI 0: C=acc ; 1: C=acc+EX(f32).  OBF 0: C f32 ; 1: C bf16.
// a_seg/c_seg != 0 remap row m -> (m/seg)*tot + off + m%seg
template <int EPI, int OBF>
__global__ void __launch_bounds__(256)
gemm_bf16_kernel(const unsigned short* __restrict__ A, const unsigned short* __restrict__ Bw,
                 void* __restrict__ Cv, const float* __restrict__ EX,
                 int M, int N, int K, int lda, int ldc,
                 int a_seg, int a_tot, int a_off,
                 int c_seg, int c_tot, int c_off) {
  __shared__ unsigned short sA[2][128 * 48];
  __shared__ unsigned short sB[2][128 * 48];
  const int tid  = threadIdx.x;
  const int lane = tid & 31, wave = tid >> 5;
  const int l16  = lane & 15, half = lane >> 4;
  const int wm   = wave >> 2, wn = wave & 3;       // 2x4 wave grid, wave tile 64x32
  const int bm   = blockIdx.y * 128, bn = blockIdx.x * 128;

  auto load_a = [&](int kt, int buf) {
    int row = tid >> 1, hf = tid & 1;              // 2 threads per 32-elem row
    int ar  = bm + row;
    UInt4x lo = {0,0,0,0}, hi = {0,0,0,0};
    if (ar < M) {
      int arow = ar;
      if (a_seg > 0) arow = (ar / a_seg) * a_tot + a_off + (ar % a_seg);
      const unsigned short* p = A + (size_t)arow * lda + kt + hf * 16;
      lo = *(const UInt4x*)p;
      hi = *(const UInt4x*)(p + 8);
    }
    *(UInt4x*)&sA[buf][row * 48 + hf * 16]     = lo;
    *(UInt4x*)&sA[buf][row * 48 + hf * 16 + 8] = hi;
  };
  auto load_b = [&](int kt, int buf) {
#if USE_TDM
    if (wave == 0)
      tdm_load_tile_128x32(Bw + (size_t)bn * K + kt, lds_off(&sB[buf][0]), K);
#else
    int row = tid >> 1, hf = tid & 1;
    int br  = bn + row;
    UInt4x lo = {0,0,0,0}, hi = {0,0,0,0};
    if (br < N) {
      const unsigned short* p = Bw + (size_t)br * K + kt + hf * 16;
      lo = *(const UInt4x*)p;
      hi = *(const UInt4x*)(p + 8);
    }
    *(UInt4x*)&sB[buf][row * 48 + hf * 16]     = lo;
    *(UInt4x*)&sB[buf][row * 48 + hf * 16 + 8] = hi;
#endif
  };

  f32x8 acc[4][2];
#pragma unroll
  for (int mi = 0; mi < 4; mi++)
#pragma unroll
    for (int ni = 0; ni < 2; ni++) acc[mi][ni] = zero8();

  load_b(0, 0);
  load_a(0, 0);
#if USE_TDM
  if (wave == 0) __builtin_amdgcn_s_wait_tensorcnt(0);
#endif
  __syncthreads();

  for (int kt = 0; kt < K; kt += 32) {
    int cur = (kt >> 5) & 1, nxt = cur ^ 1;
    if (kt + 32 < K) { load_b(kt + 32, nxt); load_a(kt + 32, nxt); }

    bf16x16 af[4], bfr[2];
#pragma unroll
    for (int mi = 0; mi < 4; mi++) {
      int base = (wm * 64 + mi * 16 + l16) * 48 + half * 8;   // A-frag pattern
      af[mi] = ld_frag(&sA[cur][base], &sA[cur][base + 16]);
    }
#pragma unroll
    for (int ni = 0; ni < 2; ni++) {
      int base = (wn * 32 + ni * 16 + l16) * 48 + half * 8;   // B-frag pattern
      bfr[ni] = ld_frag(&sB[cur][base], &sB[cur][base + 16]);
    }
#pragma unroll
    for (int mi = 0; mi < 4; mi++)
#pragma unroll
      for (int ni = 0; ni < 2; ni++)
        acc[mi][ni] = wmma_bf16(af[mi], bfr[ni], acc[mi][ni]);

#if USE_TDM
    if (wave == 0) __builtin_amdgcn_s_wait_tensorcnt(0);
#endif
    __syncthreads();
  }

  float* Cf = (float*)Cv;
  unsigned short* Cb = (unsigned short*)Cv;
#pragma unroll
  for (int mi = 0; mi < 4; mi++) {
#pragma unroll
    for (int ni = 0; ni < 2; ni++) {
#pragma unroll
      for (int r = 0; r < 8; r++) {
        int m = bm + wm * 64 + mi * 16 + r + half * 8;
        int n = bn + wn * 32 + ni * 16 + l16;
        if (m < M && n < N) {
          int crow = m;
          if (c_seg > 0) crow = (m / c_seg) * c_tot + c_off + (m % c_seg);
          size_t idx = (size_t)crow * ldc + n;
          float v = acc[mi][ni][r];
          if (EPI == 1) v += EX[idx];
          if (OBF) Cb[idx] = f2bf(v);
          else     Cf[idx] = v;
        }
      }
    }
  }
}

// ---------- Fused gate/up dual-GEMM: C = bf16( gelu(A@Wg^T) * (A@Wu^T) ) ----------
__global__ void __launch_bounds__(256)
gemm_gateup_kernel(const unsigned short* __restrict__ A,
                   const unsigned short* __restrict__ Bg,
                   const unsigned short* __restrict__ Bu,
                   unsigned short* __restrict__ C,
                   int M, int N, int K, int lda, int ldc) {
  __shared__ unsigned short sA[2][128 * 48];
  __shared__ unsigned short sG[2][128 * 48];
  __shared__ unsigned short sU[2][128 * 48];
  const int tid  = threadIdx.x;
  const int lane = tid & 31, wave = tid >> 5;
  const int l16  = lane & 15, half = lane >> 4;
  const int wm   = wave >> 2, wn = wave & 3;
  const int bm   = blockIdx.y * 128, bn = blockIdx.x * 128;

  auto load_a = [&](int kt, int buf) {
    int row = tid >> 1, hf = tid & 1;
    int ar  = bm + row;
    UInt4x lo = {0,0,0,0}, hi = {0,0,0,0};
    if (ar < M) {
      const unsigned short* p = A + (size_t)ar * lda + kt + hf * 16;
      lo = *(const UInt4x*)p;
      hi = *(const UInt4x*)(p + 8);
    }
    *(UInt4x*)&sA[buf][row * 48 + hf * 16]     = lo;
    *(UInt4x*)&sA[buf][row * 48 + hf * 16 + 8] = hi;
  };
  auto load_b2 = [&](int kt, int buf) {
#if USE_TDM
    if (wave == 0) {
      tdm_load_tile_128x32(Bg + (size_t)bn * K + kt, lds_off(&sG[buf][0]), K);
      tdm_load_tile_128x32(Bu + (size_t)bn * K + kt, lds_off(&sU[buf][0]), K);
    }
#else
    int row = tid >> 1, hf = tid & 1;
    int br  = bn + row;
    UInt4x g_lo = {0,0,0,0}, g_hi = {0,0,0,0}, u_lo = {0,0,0,0}, u_hi = {0,0,0,0};
    if (br < N) {
      const unsigned short* pg = Bg + (size_t)br * K + kt + hf * 16;
      const unsigned short* pu = Bu + (size_t)br * K + kt + hf * 16;
      g_lo = *(const UInt4x*)pg; g_hi = *(const UInt4x*)(pg + 8);
      u_lo = *(const UInt4x*)pu; u_hi = *(const UInt4x*)(pu + 8);
    }
    *(UInt4x*)&sG[buf][row * 48 + hf * 16]     = g_lo;
    *(UInt4x*)&sG[buf][row * 48 + hf * 16 + 8] = g_hi;
    *(UInt4x*)&sU[buf][row * 48 + hf * 16]     = u_lo;
    *(UInt4x*)&sU[buf][row * 48 + hf * 16 + 8] = u_hi;
#endif
  };

  f32x8 accg[4][2], accu[4][2];
#pragma unroll
  for (int mi = 0; mi < 4; mi++)
#pragma unroll
    for (int ni = 0; ni < 2; ni++) { accg[mi][ni] = zero8(); accu[mi][ni] = zero8(); }

  load_b2(0, 0);
  load_a(0, 0);
#if USE_TDM
  if (wave == 0) __builtin_amdgcn_s_wait_tensorcnt(0);
#endif
  __syncthreads();

  for (int kt = 0; kt < K; kt += 32) {
    int cur = (kt >> 5) & 1, nxt = cur ^ 1;
    if (kt + 32 < K) { load_b2(kt + 32, nxt); load_a(kt + 32, nxt); }

    bf16x16 af[4], bg[2], bu[2];
#pragma unroll
    for (int mi = 0; mi < 4; mi++) {
      int base = (wm * 64 + mi * 16 + l16) * 48 + half * 8;
      af[mi] = ld_frag(&sA[cur][base], &sA[cur][base + 16]);
    }
#pragma unroll
    for (int ni = 0; ni < 2; ni++) {
      int base = (wn * 32 + ni * 16 + l16) * 48 + half * 8;
      bg[ni] = ld_frag(&sG[cur][base], &sG[cur][base + 16]);
      bu[ni] = ld_frag(&sU[cur][base], &sU[cur][base + 16]);
    }
#pragma unroll
    for (int mi = 0; mi < 4; mi++)
#pragma unroll
      for (int ni = 0; ni < 2; ni++) {
        accg[mi][ni] = wmma_bf16(af[mi], bg[ni], accg[mi][ni]);
        accu[mi][ni] = wmma_bf16(af[mi], bu[ni], accu[mi][ni]);
      }

#if USE_TDM
    if (wave == 0) __builtin_amdgcn_s_wait_tensorcnt(0);
#endif
    __syncthreads();
  }

#pragma unroll
  for (int mi = 0; mi < 4; mi++) {
#pragma unroll
    for (int ni = 0; ni < 2; ni++) {
#pragma unroll
      for (int r = 0; r < 8; r++) {
        int m = bm + wm * 64 + mi * 16 + r + half * 8;
        int n = bn + wn * 32 + ni * 16 + l16;
        if (m < M && n < N)
          C[(size_t)m * ldc + n] = f2bf(gelu_tanh(accg[mi][ni][r]) * accu[mi][ni][r]);
      }
    }
  }
}

// ---------- RoPE in place on bf16 (H=8 for Q, H=1 for K) ----------
__global__ void __launch_bounds__(256)
rope_kernel(unsigned short* __restrict__ buf, const int* __restrict__ pos_ids,
            int B_, int S_, int H_) {
  size_t idx = (size_t)blockIdx.x * blockDim.x + threadIdx.x;
  size_t total = (size_t)B_ * S_ * H_ * 128;
  if (idx >= total) return;
  int i = (int)(idx & 127);
  int h = (int)((idx >> 7) % H_);
  int t = (int)((idx / ((size_t)H_ * 128)) % S_);
  int b = (int)(idx / ((size_t)S_ * H_ * 128));
  int p = pos_ids[b * S_ + t];
  float invf = __expf(-(float)(2 * i) * (1.0f / 256.0f) * 9.210340371976184f); // ln(1e4)
  float ang = (float)p * invf;
  float sn, cs;
  __sincosf(ang, &sn, &cs);
  size_t base = ((size_t)(b * S_ + t) * H_ + h) * 256;
  float q0 = bf2f(buf[base + i]), q1 = bf2f(buf[base + 128 + i]);
  buf[base + i]       = f2bf(q0 * cs - q1 * sn);
  buf[base + 128 + i] = f2bf(q1 * cs + q0 * sn);
}

// ---------- V transpose: (B,S,256) bf16 -> (B,256,SP) bf16, zero-padded ----------
__global__ void __launch_bounds__(256)
vtrans_kernel(const unsigned short* __restrict__ v, unsigned short* __restrict__ vt,
              int B_, int S_, int SP_) {
  size_t idx = (size_t)blockIdx.x * blockDim.x + threadIdx.x;
  size_t total = (size_t)B_ * 256 * SP_;
  if (idx >= total) return;
  int t = (int)(idx % SP_);
  int d = (int)((idx / SP_) % 256);
  int b = (int)(idx / ((size_t)SP_ * 256));
  vt[idx] = (t < S_) ? v[((size_t)(b * S_ + t)) * 256 + d] : (unsigned short)0;
}

// ---------- Flash attention (MQA): 1 wave = 16 q rows, full HD=256, bf16 out ----------
__global__ void __launch_bounds__(128)
attention_kernel(const unsigned short* __restrict__ qbf,
                 const unsigned short* __restrict__ kbf,
                 const unsigned short* __restrict__ vtbf,
                 const float* __restrict__ mask,
                 unsigned short* __restrict__ att_out, int B_, int S_, int SP_) {
  const int lane = threadIdx.x & 31;
  const int wave = threadIdx.x >> 5;
  const int l16  = lane & 15, half = lane >> 4;
  const int b = blockIdx.z, h = blockIdx.y;
  const int q0 = blockIdx.x * 64 + wave * 16;

  __shared__ unsigned short psh[4 * 16 * 48];   // per-wave P transpose staging
  unsigned short* lp = psh + wave * 16 * 48;

  bf16x16 qa[8];
  {
    int qrow = q0 + l16; if (qrow >= S_) qrow = S_ - 1;
    const unsigned short* qbase = qbf + ((size_t)(b * S_ + qrow)) * 2048 + h * 256;
#pragma unroll
    for (int kc = 0; kc < 8; kc++)
      qa[kc] = ld_frag(qbase + kc * 32 + half * 8, qbase + kc * 32 + half * 8 + 16);
  }

  f32x8 o[16];
#pragma unroll
  for (int i = 0; i < 16; i++) o[i] = zero8();
  float rmax[8], rsum[8];
#pragma unroll
  for (int r = 0; r < 8; r++) { rmax[r] = -3.0e38f; rsum[r] = 0.0f; }

  const float scaling = 0.0625f;  // 256^-0.5

  for (int kv0 = 0; kv0 < S_; kv0 += 32) {
    f32x8 s0 = zero8(), s1 = zero8();
    int kr0 = kv0 + l16;      if (kr0 >= S_) kr0 = S_ - 1;
    int kr1 = kv0 + 16 + l16; if (kr1 >= S_) kr1 = S_ - 1;
    const unsigned short* kp0 = kbf + ((size_t)(b * S_ + kr0)) * 256;
    const unsigned short* kp1 = kbf + ((size_t)(b * S_ + kr1)) * 256;
#pragma unroll
    for (int kc = 0; kc < 8; kc++) {
      bf16x16 b0 = ld_frag(kp0 + kc * 32 + half * 8, kp0 + kc * 32 + half * 8 + 16);
      bf16x16 b1 = ld_frag(kp1 + kc * 32 + half * 8, kp1 + kc * 32 + half * 8 + 16);
      s0 = wmma_bf16(qa[kc], b0, s0);
      s1 = wmma_bf16(qa[kc], b1, s1);
    }

    float p0[8], p1[8], sc[8];
#pragma unroll
    for (int r = 0; r < 8; r++) {
      int m  = q0 + r + half * 8;
      int mc = m < S_ ? m : S_ - 1;
      int n0 = kv0 + l16, n1 = kv0 + 16 + l16;
      int n0c = n0 < S_ ? n0 : S_ - 1;
      int n1c = n1 < S_ ? n1 : S_ - 1;
      float v0 = s0[r] * scaling + mask[((size_t)b * S_ + mc) * S_ + n0c];
      float v1 = s1[r] * scaling + mask[((size_t)b * S_ + mc) * S_ + n1c];
      if (n0 >= S_) v0 = -3.0e38f;
      if (n1 >= S_) v1 = -3.0e38f;
      float cm = fmaxf(v0, v1);
#pragma unroll
      for (int off = 1; off < 16; off <<= 1) cm = fmaxf(cm, __shfl_xor(cm, off, 32));
      float nm = fmaxf(rmax[r], cm);
      sc[r] = __expf(rmax[r] - nm);
      rmax[r] = nm;
      p0[r] = __expf(v0 - nm);
      p1[r] = __expf(v1 - nm);
      float cs = p0[r] + p1[r];
#pragma unroll
      for (int off = 1; off < 16; off <<= 1) cs += __shfl_xor(cs, off, 32);
      rsum[r] = rsum[r] * sc[r] + cs;
    }
#pragma unroll
    for (int i = 0; i < 16; i++)
#pragma unroll
      for (int r = 0; r < 8; r++) o[i][r] *= sc[r];

#pragma unroll
    for (int r = 0; r < 8; r++) {
      lp[(r + half * 8) * 48 + l16]      = f2bf(p0[r]);
      lp[(r + half * 8) * 48 + 16 + l16] = f2bf(p1[r]);
    }
    bf16x16 pa = ld_frag(lp + l16 * 48 + half * 8, lp + l16 * 48 + half * 8 + 16);

    const unsigned short* vb = vtbf + ((size_t)b * 256) * (size_t)SP_ + kv0;
#pragma unroll
    for (int hi = 0; hi < 16; hi++) {
      const unsigned short* vp = vb + (size_t)(hi * 16 + l16) * SP_ + half * 8;
      o[hi] = wmma_bf16(pa, ld_frag(vp, vp + 16), o[hi]);
    }
  }

#pragma unroll
  for (int hi = 0; hi < 16; hi++) {
#pragma unroll
    for (int r = 0; r < 8; r++) {
      int m = q0 + r + half * 8;
      if (m < S_)
        att_out[((size_t)(b * S_ + m)) * 2048 + h * 256 + hi * 16 + l16] =
            f2bf(o[hi][r] / rsum[r]);
    }
  }
}

// ---------- host orchestration ----------
extern "C" void kernel_launch(void* const* d_in, const int* in_sizes, int n_in,
                              void* d_out, int out_size, void* d_ws, size_t ws_size,
                              hipStream_t stream) {
  (void)in_sizes; (void)n_in; (void)out_size; (void)ws_size;
  const float* xs[3] = {(const float*)d_in[0], (const float*)d_in[1], (const float*)d_in[2]};
  const float* mask  = (const float*)d_in[3];
  const int*   pos   = (const int*)d_in[4];
  const float* prm[3][9];
  for (int e = 0; e < 3; e++)
    for (int j = 0; j < 9; j++) prm[e][j] = (const float*)d_in[5 + e * 9 + j];
  enum { LN_IN = 0, WQ, WK, WV, WO, LN_POST, WG, WU, WD };

  const int B = 2, S = 1635, SP = 1664;
  const int Sseg[3] = {1072, 512, 51};
  const int dss[3]  = {2048, 1024, 1024};
  const int mss[3]  = {16384, 4096, 4096};
  const int soff[3] = {0, 1072, 1584};
  const int Mr[3]   = {2144, 1024, 102};

  char* w = (char*)d_ws;
  size_t off = 0;
  auto take = [&](size_t elems, size_t esz) -> char* {
    char* p = w + off;
    off += ((elems * esz + 255) & ~(size_t)255);
    return p;
  };

  // bf16 weight mirrors (converted once per launch)
  unsigned short* wbf[3][9];
  const int wjs[7] = {WQ, WK, WV, WO, WG, WU, WD};
  for (int e = 0; e < 3; e++) {
    int d = dss[e], m = mss[e];
    size_t wsz[9];
    wsz[WQ] = (size_t)2048 * d; wsz[WK] = (size_t)256 * d; wsz[WV] = (size_t)256 * d;
    wsz[WO] = (size_t)d * 2048; wsz[WG] = (size_t)m * d;   wsz[WU] = (size_t)m * d;
    wsz[WD] = (size_t)d * m;
    for (int jj = 0; jj < 7; jj++) {
      int j = wjs[jj];
      wbf[e][j] = (unsigned short*)take(wsz[j], 2);
    }
  }
  unsigned short* hb   = (unsigned short*)take((size_t)2144 * 2048, 2);
  unsigned short* qb   = (unsigned short*)take((size_t)B * S * 2048, 2);
  unsigned short* kb   = (unsigned short*)take((size_t)B * S * 256, 2);
  unsigned short* vb   = (unsigned short*)take((size_t)B * S * 256, 2);
  unsigned short* vt   = (unsigned short*)take((size_t)B * 256 * SP, 2);
  unsigned short* attb = (unsigned short*)take((size_t)B * S * 2048, 2);
  unsigned short* midb = (unsigned short*)take((size_t)2144 * 16384, 2);

  float* outp[3];
  outp[0] = (float*)d_out;
  outp[1] = outp[0] + (size_t)Mr[0] * dss[0];
  outp[2] = outp[1] + (size_t)Mr[1] * dss[1];

  auto gg = [](int M, int N) { return dim3((unsigned)((N + 127) / 128), (unsigned)((M + 127) / 128), 1); };

  // Phase 0: convert weights to bf16
  for (int e = 0; e < 3; e++) {
    int d = dss[e], m = mss[e];
    size_t wsz[9];
    wsz[WQ] = (size_t)2048 * d; wsz[WK] = (size_t)256 * d; wsz[WV] = (size_t)256 * d;
    wsz[WO] = (size_t)d * 2048; wsz[WG] = (size_t)m * d;   wsz[WU] = (size_t)m * d;
    wsz[WD] = (size_t)d * m;
    for (int jj = 0; jj < 7; jj++) {
      int j = wjs[jj];
      cvt_kernel<<<2048, 256, 0, stream>>>(prm[e][j], wbf[e][j], wsz[j]);
    }
  }

  // Phase 1: per-expert pre-norm + QKV projections (bf16 out, rows remapped into (B,S,*))
  for (int e = 0; e < 3; e++) {
    int M = Mr[e], d = dss[e];
    rmsnorm_kernel<<<M, 256, 0, stream>>>(xs[e], prm[e][LN_IN], hb, M, d);
    gemm_bf16_kernel<0, 1><<<gg(M, 2048), 256, 0, stream>>>(hb, wbf[e][WQ], qb, nullptr,
        M, 2048, d, d, 2048, 0, 0, 0, Sseg[e], S, soff[e]);
    gemm_bf16_kernel<0, 1><<<gg(M, 256), 256, 0, stream>>>(hb, wbf[e][WK], kb, nullptr,
        M, 256, d, d, 256, 0, 0, 0, Sseg[e], S, soff[e]);
    gemm_bf16_kernel<0, 1><<<gg(M, 256), 256, 0, stream>>>(hb, wbf[e][WV], vb, nullptr,
        M, 256, d, d, 256, 0, 0, 0, Sseg[e], S, soff[e]);
  }

  // Phase 2: RoPE in place, V transpose, fused attention (bf16 out)
  rope_kernel<<<(B * S * 8 * 128) / 256, 256, 0, stream>>>(qb, pos, B, S, 8);
  rope_kernel<<<(B * S * 1 * 128) / 256, 256, 0, stream>>>(kb, pos, B, S, 1);
  vtrans_kernel<<<(B * 256 * SP) / 256, 256, 0, stream>>>(vb, vt, B, S, SP);
  attention_kernel<<<dim3((S + 63) / 64, 8, B), 128, 0, stream>>>(qb, kb, vt, mask, attb, B, S, SP);

  // Phase 3: output proj + residual, post-norm, fused gate*up, down proj + residual
  for (int e = 0; e < 3; e++) {
    int M = Mr[e], d = dss[e], m = mss[e];
    gemm_bf16_kernel<1, 0><<<gg(M, d), 256, 0, stream>>>(attb, wbf[e][WO], outp[e], xs[e],
        M, d, 2048, 2048, d, Sseg[e], S, soff[e], 0, 0, 0);
    rmsnorm_kernel<<<M, 256, 0, stream>>>(outp[e], prm[e][LN_POST], hb, M, d);
    gemm_gateup_kernel<<<gg(M, m), 256, 0, stream>>>(hb, wbf[e][WG], wbf[e][WU], midb,
        M, m, d, d, m);
    gemm_bf16_kernel<1, 0><<<gg(M, d), 256, 0, stream>>>(midb, wbf[e][WD], outp[e], outp[e],
        M, d, m, m, d, 0, 0, 0, 0, 0, 0);
  }
}
